// GCN_18476949307803
// MI455X (gfx1250) — matrix operand order repaired and checked by reading, not compiled, not run on hardware
//
#include <hip/hip_runtime.h>
#include <stdint.h>

// ---------------------------------------------------------------------------
// GCN layer for MI455X (gfx1250, wave32, WMMA):
//   h   = seq @ W^T          [8192x512]*[512x256]
//   agg = adj @ h            [8192x8192]*[8192x256]
//   out = relu(agg + bias)
// bf16 WMMA, f32 accumulate. adj (268 MB) streams from HBM exactly once via
// double-buffered global_load_async_to_lds_b128; hT (4 MB bf16) is L2-resident.
// ---------------------------------------------------------------------------

typedef __bf16 bf16_t;
typedef __attribute__((ext_vector_type(16))) __bf16        v16bf;
typedef __attribute__((ext_vector_type(8)))  float         v8f;
typedef __attribute__((ext_vector_type(4)))  float         v4f;
typedef __attribute__((ext_vector_type(4)))  unsigned int  v4u;

#define SEQ_N   8192
#define IN_FT   512
#define OUT_FT  256

// packed f32x2 -> bf16x2 (RNE), one VALU op for two elements
static __device__ __forceinline__ unsigned cvt_pk_bf16(float lo, float hi) {
    unsigned r;
    asm("v_cvt_pk_bf16_f32 %0, %1, %2" : "=v"(r) : "v"(lo), "v"(hi));
    return r;
}

union Frag { v16bf v; unsigned d[8]; };

// Pack 16 f32 (four contiguous float4 runs, already in ISA element order)
// into a 16x32 A / 32x16 B bf16 fragment.
static __device__ __forceinline__ v16bf pack16(v4f c0, v4f c1, v4f c2, v4f c3) {
    Frag r;
    r.d[0] = cvt_pk_bf16(c0[0], c0[1]);
    r.d[1] = cvt_pk_bf16(c0[2], c0[3]);
    r.d[2] = cvt_pk_bf16(c1[0], c1[1]);
    r.d[3] = cvt_pk_bf16(c1[2], c1[3]);
    r.d[4] = cvt_pk_bf16(c2[0], c2[1]);
    r.d[5] = cvt_pk_bf16(c2[2], c2[3]);
    r.d[6] = cvt_pk_bf16(c3[0], c3[1]);
    r.d[7] = cvt_pk_bf16(c3[2], c3[3]);
    return r.v;
}

// ---------------------------------------------------------------------------
// Kernel 1: h = seq @ W^T, stored transposed bf16: hT[n][m] (n<256, m<8192).
// One 16x16 tile per wave; 8 waves/block -> 1024 blocks.
// A layout (16x32 bf16): lane&15 = M, half = lane>>4,
//   e=0..7 -> K = 8*half+e ; e=8..15 -> K = 16+8*half+(e-8)  (two 32B runs)
// B layout (32x16 bf16): lane&15 = N, same K mapping as A     (two 32B runs;
//   B[k][n] = W[n][k] so W rows are already K-contiguous)
// ---------------------------------------------------------------------------
__global__ __launch_bounds__(256) void gcn_fc1_kernel(
    const float* __restrict__ seq, const float* __restrict__ W,
    bf16_t* __restrict__ hT)
{
    const int lane = threadIdx.x & 31;
    const int wave = threadIdx.x >> 5;
    const int g    = blockIdx.x * 8 + wave;   // 0..8191 tiles
    const int n0   = (g & 15) * 16;
    const int m0   = (g >> 4) * 16;
    const int nn   = lane & 15;
    const int hs   = lane >> 4;

    v8f acc = {};
#pragma unroll 4
    for (int k = 0; k < IN_FT; k += 32) {
        const float* pa = seq + (size_t)(m0 + nn) * IN_FT + k + 8 * hs;
        v16bf a = pack16(*(const v4f*)(pa),      *(const v4f*)(pa + 4),
                         *(const v4f*)(pa + 16), *(const v4f*)(pa + 20));
        const float* pb = W + (size_t)(n0 + nn) * IN_FT + k + 8 * hs;
        v16bf b = pack16(*(const v4f*)(pb),      *(const v4f*)(pb + 4),
                         *(const v4f*)(pb + 16), *(const v4f*)(pb + 20));
        acc = __builtin_amdgcn_wmma_f32_16x16x32_bf16(
            false, a, false, b, (short)0, acc, false, false);
    }

    // C/D: VGPR v -> M = m0 + 8*half + v, N = n0 + (lane&15).
    // 8 contiguous-M bf16 -> one 16B store per lane into hT[n][m].
    union { v4u v; unsigned d[4]; } pk;
#pragma unroll
    for (int i = 0; i < 4; ++i) pk.d[i] = cvt_pk_bf16(acc[2 * i], acc[2 * i + 1]);
    *(v4u*)(hT + (size_t)(n0 + nn) * SEQ_N + m0 + 8 * hs) = pk.v;
}

// ---------------------------------------------------------------------------
// Kernel 2: out = relu(adj @ h + bias).
// Block: 256 thr = 8 waves (2M x 4N); block tile 32x256, wave tile 16x64.
// Grid: 8192/32 = 256 blocks. K in steps of 64, double-buffered async->LDS.
// ---------------------------------------------------------------------------
__global__ __launch_bounds__(256) void gcn_agg_kernel(
    const float* __restrict__ adj, const bf16_t* __restrict__ hT,
    const float* __restrict__ bias, float* __restrict__ out)
{
    __shared__ __align__(16) float tileA[2][32 * 64];   // 2 x 8KB f32 adj tiles

    const int tid   = threadIdx.x;
    const int lane  = tid & 31;
    const int wave  = tid >> 5;
    const int wm    = wave >> 2;              // 0..1  (M wave)
    const int wn    = wave & 3;               // 0..3  (N wave)
    const int m_blk = blockIdx.x * 32;
    const int n0    = wn * 64;
    const int nn    = lane & 15;
    const int hs    = lane >> 4;

    // async tile fetch: thread -> (row, 16B chunk); 2 chunks 128B apart cover
    // the 256B row segment. LDS offset = low 32 bits of generic shared ptr.
    const int row    = tid >> 3;              // 0..31
    const int cchunk = tid & 7;               // 0..7
    const unsigned ldsBase = (unsigned)(uintptr_t)(&tileA[0][0]);
    const unsigned ldsOff  = ldsBase + (unsigned)(((row << 6) + (cchunk << 2)) << 2);
    const float* gRow = adj + (size_t)(m_blk + row) * SEQ_N + (cchunk << 2);

    { // prologue: buffer 0 <- K chunk [0,64)
        unsigned l0 = ldsOff;
        unsigned long long ga = (unsigned long long)(uintptr_t)gRow;
        asm volatile("global_load_async_to_lds_b128 %0, %1, off"
                     :: "v"(l0), "v"(ga) : "memory");
        asm volatile("global_load_async_to_lds_b128 %0, %1, off offset:128"
                     :: "v"(l0), "v"(ga) : "memory");
    }

    v8f acc[4] = {v8f{}, v8f{}, v8f{}, v8f{}};
    const int mfr = wm * 16 + nn;             // A-frag row within the 32-row tile

    int buf = 0;
    for (int kb = 0; kb < SEQ_N; kb += 64, buf ^= 1) {
        if (kb + 64 < SEQ_N) {                // issue next tile into other buffer
            unsigned l0 = ldsOff + (unsigned)((buf ^ 1) * (32 * 64 * 4));
            unsigned long long ga = (unsigned long long)(uintptr_t)(gRow + kb + 64);
            asm volatile("global_load_async_to_lds_b128 %0, %1, off"
                         :: "v"(l0), "v"(ga) : "memory");
            asm volatile("global_load_async_to_lds_b128 %0, %1, off offset:128"
                         :: "v"(l0), "v"(ga) : "memory");
            asm volatile("s_wait_asynccnt 0x2" ::: "memory");  // current tile landed
        } else {
            asm volatile("s_wait_asynccnt 0x0" ::: "memory");
        }
        __syncthreads();

        const float* tA = &tileA[buf][0];
#pragma unroll
        for (int ks = 0; ks < 64; ks += 32) {
            const float* p = tA + mfr * 64 + ks + 16 * hs;    // ds_load_b128 x4
            v16bf a = pack16(*(const v4f*)(p),      *(const v4f*)(p + 4),
                             *(const v4f*)(p + 8),  *(const v4f*)(p + 12));
#pragma unroll
            for (int t = 0; t < 4; ++t) {
                v16bf b = *(const v16bf*)(hT + (size_t)(n0 + t * 16 + nn) * SEQ_N
                                          + kb + ks + 16 * hs);
                acc[t] = __builtin_amdgcn_wmma_f32_16x16x32_bf16(
                    false, a, false, b, (short)0, acc[t], false, false);
            }
        }
        __syncthreads();                      // all reads done before reuse
    }

    // epilogue: bias + relu, f32 store. M = m_blk + wm*16 + 8*half + v
#pragma unroll
    for (int t = 0; t < 4; ++t) {
        const int n  = n0 + t * 16 + nn;
        const float bz = bias[n];
#pragma unroll
        for (int v = 0; v < 8; ++v) {
            float r = acc[t][v] + bz;
            out[(size_t)(m_blk + wm * 16 + 8 * hs + v) * OUT_FT + n] = fmaxf(r, 0.0f);
        }
    }
}

// ---------------------------------------------------------------------------
extern "C" void kernel_launch(void* const* d_in, const int* in_sizes, int n_in,
                              void* d_out, int out_size, void* d_ws, size_t ws_size,
                              hipStream_t stream) {
    (void)in_sizes; (void)n_in; (void)out_size; (void)ws_size;
    const float* seq  = (const float*)d_in[0];   // [8192,512]
    const float* adj  = (const float*)d_in[1];   // [8192,8192]
    const float* W    = (const float*)d_in[2];   // [256,512]
    const float* bias = (const float*)d_in[3];   // [256]
    float*       out  = (float*)d_out;           // [8192,256]
    bf16_t*      hT   = (bf16_t*)d_ws;           // [256,8192] bf16 = 4 MB

    gcn_fc1_kernel<<<1024, 256, 0, stream>>>(seq, W, hT);
    gcn_agg_kernel<<<256, 256, 0, stream>>>(adj, hT, bias, out);
}
